// ImprovedLatticeRNNCell_12275016532185
// MI455X (gfx1250) — matrix operand order repaired and checked by reading, not compiled, not run on hardware
//
#include <hip/hip_runtime.h>

typedef __attribute__((ext_vector_type(16))) _Float16 v16h;
typedef __attribute__((ext_vector_type(8)))  float    v8f;
typedef __attribute__((ext_vector_type(2)))  _Float16 h2;
typedef __attribute__((ext_vector_type(4)))  _Float16 h4;
typedef __attribute__((ext_vector_type(4)))  int      v4i;

// ---- gfx1250 async global->LDS copy (ASYNCcnt) with fallback ----
#if defined(__has_builtin)
# if __has_builtin(__builtin_amdgcn_global_load_async_to_lds_b128)
#  define USE_ASYNC_LDS 1
# endif
#endif
#ifndef USE_ASYNC_LDS
# define USE_ASYNC_LDS 0
#endif

__device__ __forceinline__ void cp128_to_lds(_Float16* l, const _Float16* g){
#if USE_ASYNC_LDS
  __builtin_amdgcn_global_load_async_to_lds_b128(
      (__attribute__((address_space(1))) v4i*)(uintptr_t)g,
      (__attribute__((address_space(3))) v4i*)(uintptr_t)(void*)l,
      0, 0);
#else
  *(uint4*)l = *(const uint4*)g;
#endif
}

__device__ __forceinline__ void wait_lds_copies(){
#if USE_ASYNC_LDS
# if __has_builtin(__builtin_amdgcn_s_wait_asynccnt)
  __builtin_amdgcn_s_wait_asynccnt(0);
# else
  asm volatile("s_wait_asynccnt 0x0" ::: "memory");
# endif
#endif
}

__device__ __forceinline__ float sigmoidf_(float x){ return 1.0f/(1.0f + __expf(-x)); }

// ---------------- small helper kernels ----------------

__global__ void k_cvt(const float* __restrict__ s, _Float16* __restrict__ d, int n){
  int i = blockIdx.x*blockDim.x + threadIdx.x;
  if (i < n) d[i] = (_Float16)s[i];
}

// interleave hidden_left / hidden_up -> sp16 [2B, H]
__global__ void k_sp(const float* __restrict__ hl, const float* __restrict__ hu,
                     _Float16* __restrict__ sp, int BH, int H){
  int i = blockIdx.x*blockDim.x + threadIdx.x;
  if (i >= BH) return;
  int b = i / H, h = i - b*H;
  sp[(size_t)(2*b)*H   + h] = (_Float16)hl[i];
  sp[(size_t)(2*b+1)*H + h] = (_Float16)hu[i];
}

// gate_in = [hidden_prev, spatial_combined, x_proj]  -> [B, 3H] f16
__global__ void k_concat(const float* __restrict__ hp, const _Float16* __restrict__ spc,
                         const _Float16* __restrict__ xp, _Float16* __restrict__ g,
                         int BH, int H){
  int i = blockIdx.x*blockDim.x + threadIdx.x;
  if (i >= BH) return;
  int b = i / H, h = i - b*H;
  size_t o = (size_t)b*3*H + h;
  g[o]       = (_Float16)hp[i];
  g[o + H]   = spc[i];
  g[o + 2*H] = xp[i];
}

// ---------------- WMMA GEMM: out = A[M,K] @ W[N,K]^T + bias ----------------
// Block tile 64(M) x 128(N), BK=32, 8 waves in 4x2; each wave: 16x64 strip = 4 WMMAs/k-step.
// Double-buffered LDS, one barrier per k-step, async global->LDS copies (ASYNCcnt).
// EPI 0: store f16
// EPI 1: average row pairs (2b,2b+1) -> out row b (attention out-proj + positional mean)
// EPI 2: gate fuse: out = hp + sigmoid(acc+bias) * spc

#define GBM 64
#define GBN 128
#define GBK 32
#define LPAD 16
#define LSTR (GBK + LPAD)   // 48 halves = 96B rows -> every 16B chunk aligned

template<int EPI>
__global__ __launch_bounds__(256) void k_gemm(
    const _Float16* __restrict__ A, const _Float16* __restrict__ W,
    const float* __restrict__ bias, _Float16* __restrict__ out,
    const float* __restrict__ hp, const _Float16* __restrict__ spc,
    int M, int N, int K)
{
  __shared__ _Float16 As[2][GBM][LSTR];   // 12 KB
  __shared__ _Float16 Bs[2][GBN][LSTR];   // 24 KB

  const int tid   = threadIdx.x;
  const int lane  = tid & 31;
  const int wave  = tid >> 5;
  const int waveM = wave & 3;        // 4 row tiles of 16
  const int waveN = wave >> 2;       // 2 col strips of 64
  const int rowBase = blockIdx.y * GBM;
  const int colBase = blockIdx.x * GBN;

  const int l15 = lane & 15;
  const int hi  = lane >> 4;
  const int arow = waveM*16 + l15;

  // cooperative-copy coordinates (whole block covers the tile)
  const int ar = tid >> 2;               // 0..63   (8 halves each)
  const int ac = (tid & 3) << 3;         // 0,8,16,24
  const int br = tid >> 1;               // 0..127  (16 halves each)
  const int bc = (tid & 1) << 4;         // 0,16

  const _Float16* gA = A + (size_t)(rowBase + ar)*K + ac;
  const _Float16* gB = W + (size_t)(colBase + br)*K + bc;

  v8f acc[4] = {};

  const int nk = K / GBK;

  // prologue: stage k-step 0 into buffer 0
  cp128_to_lds(&As[0][ar][ac], gA);
  cp128_to_lds(&Bs[0][br][bc],     gB);
  cp128_to_lds(&Bs[0][br][bc + 8], gB + 8);

  for (int kk = 0; kk < nk; ++kk) {
    wait_lds_copies();
    __syncthreads();

    if (kk + 1 < nk) {
      int nb = (kk + 1) & 1;
      int ko = (kk + 1) * GBK;
      cp128_to_lds(&As[nb][ar][ac], gA + ko);
      cp128_to_lds(&Bs[nb][br][bc],     gB + ko);
      cp128_to_lds(&Bs[nb][br][bc + 8], gB + ko + 8);
    }

    const int buf = kk & 1;

    // fetch A fragment and all four B fragments first, then chain the WMMAs
    v16h af;
    #pragma unroll
    for (int j = 0; j < 8; ++j) {
      int ka = ((j >> 2) << 4) + ((j & 3) << 1) + (hi << 3);
      ((h2*)&af)[j] = *(const h2*)&As[buf][arow][ka];
    }
    v16h bf[4];
    #pragma unroll
    for (int t = 0; t < 4; ++t) {
      const int bcol = waveN*64 + t*16 + l15;
      #pragma unroll
      for (int j = 0; j < 8; ++j) {
        int kb = ((j >> 2) << 4) + ((j & 3) << 1) + (hi << 3);
        ((h2*)&bf[t])[j] = *(const h2*)&Bs[buf][bcol][kb];
      }
    }
    #pragma unroll
    for (int t = 0; t < 4; ++t) {
      acc[t] = __builtin_amdgcn_wmma_f32_16x16x32_f16(false, af, false, bf[t],
                                                      (short)0, acc[t], false, false);
    }
    // no trailing barrier needed: next iteration writes the other buffer,
    // and a wave reaches the top barrier only after its reads completed.
  }

  #pragma unroll
  for (int t = 0; t < 4; ++t) {
    const int col = colBase + waveN*64 + t*16 + l15;
    const float bv = bias[col];
    if (EPI == 1) {
      #pragma unroll
      for (int r = 0; r < 8; r += 2) {
        int row = rowBase + waveM*16 + r + (hi << 3);    // even; pair in-lane
        float v = 0.5f*(acc[t][r] + acc[t][r+1]) + bv;
        out[(size_t)(row >> 1)*N + col] = (_Float16)v;
      }
    } else {
      #pragma unroll
      for (int r = 0; r < 8; ++r) {
        int row = rowBase + waveM*16 + r + (hi << 3);
        float v = acc[t][r] + bv;
        size_t o = (size_t)row*N + col;
        if (EPI == 2) {
          float g = sigmoidf_(v);
          v = hp[o] + g * (float)spc[o];                 // enhanced_prev
        }
        out[o] = (_Float16)v;
      }
    }
  }
}

// ---------------- tiny 2x2 multi-head attention, one wave per (b, head) ----------------
__global__ __launch_bounds__(256) void k_attn(const _Float16* __restrict__ qkv,
                                              _Float16* __restrict__ ctx, int nb){
  int gw = (blockIdx.x * 256 + threadIdx.x) >> 5;
  int lane = threadIdx.x & 31;
  int b = gw >> 2, h = gw & 3;
  if (b >= nb) return;
  const int QN = 1536, HH = 512;
  size_t r0 = (size_t)(2*b) * QN, r1 = r0 + QN;
  int d = h*128 + lane*4;                  // 4 head-dims per lane, HD=128
  h4 q0 = *(const h4*)(qkv + r0 + d);
  h4 q1 = *(const h4*)(qkv + r1 + d);
  h4 k0 = *(const h4*)(qkv + r0 + 512 + d);
  h4 k1 = *(const h4*)(qkv + r1 + 512 + d);
  h4 v0 = *(const h4*)(qkv + r0 + 1024 + d);
  h4 v1 = *(const h4*)(qkv + r1 + 1024 + d);
  float s00=0.f,s01=0.f,s10=0.f,s11=0.f;
  #pragma unroll
  for (int i=0;i<4;++i){
    float a=(float)q0[i], bq=(float)q1[i], c=(float)k0[i], e=(float)k1[i];
    s00 += a*c; s01 += a*e; s10 += bq*c; s11 += bq*e;
  }
  #pragma unroll
  for (int m=16;m>=1;m>>=1){
    s00 += __shfl_xor(s00,m,32);
    s01 += __shfl_xor(s01,m,32);
    s10 += __shfl_xor(s10,m,32);
    s11 += __shfl_xor(s11,m,32);
  }
  const float sc = 0.08838834764831845f;   // 1/sqrt(128)
  s00*=sc; s01*=sc; s10*=sc; s11*=sc;
  float m0=fmaxf(s00,s01), m1=fmaxf(s10,s11);
  float e00=__expf(s00-m0), e01=__expf(s01-m0);
  float e10=__expf(s10-m1), e11=__expf(s11-m1);
  float a00=e00/(e00+e01), a01=1.f-a00;
  float a10=e10/(e10+e11), a11=1.f-a10;
  h4 c0, c1;
  #pragma unroll
  for (int i=0;i<4;++i){
    float va=(float)v0[i], vb=(float)v1[i];
    c0[i] = (_Float16)(a00*va + a01*vb);
    c1[i] = (_Float16)(a10*va + a11*vb);
  }
  *(h4*)(ctx + (size_t)(2*b)*HH   + d) = c0;
  *(h4*)(ctx + (size_t)(2*b+1)*HH + d) = c1;
}

// ---------------- final GRU elementwise ----------------
__global__ void k_gru(const _Float16* __restrict__ gi, const _Float16* __restrict__ gh,
                      const _Float16* __restrict__ enh, float* __restrict__ out, int BH){
  int i = blockIdx.x*blockDim.x + threadIdx.x;
  if (i >= BH) return;
  int b = i >> 9, hcol = i & 511;
  size_t o = (size_t)b*1536 + hcol;
  float ir=(float)gi[o], iz=(float)gi[o+512], in_=(float)gi[o+1024];
  float hr=(float)gh[o], hz=(float)gh[o+512], hn=(float)gh[o+1024];
  float r = sigmoidf_(ir+hr);
  float z = sigmoidf_(iz+hz);
  float n = tanhf(in_ + r*hn);
  out[i] = (1.f - z)*n + z*(float)enh[i];
}

// ---------------- launch ----------------
extern "C" void kernel_launch(void* const* d_in, const int* in_sizes, int n_in,
                              void* d_out, int out_size, void* d_ws, size_t ws_size,
                              hipStream_t stream) {
  (void)in_sizes; (void)n_in; (void)out_size; (void)ws_size;
  const int B = 16384, IN = 128, H = 512, H3 = 1536;
  const float* x   = (const float*)d_in[0];
  const float* hl  = (const float*)d_in[1];
  const float* hu  = (const float*)d_in[2];
  const float* hp  = (const float*)d_in[3];
  const float* Win = (const float*)d_in[4];
  const float* bin = (const float*)d_in[5];
  const float* Aiw = (const float*)d_in[6];
  const float* Aib = (const float*)d_in[7];
  const float* Aow = (const float*)d_in[8];
  const float* Aob = (const float*)d_in[9];
  const float* Wg  = (const float*)d_in[10];
  const float* bg  = (const float*)d_in[11];
  const float* Wih = (const float*)d_in[12];
  const float* Whh = (const float*)d_in[13];
  const float* bih = (const float*)d_in[14];
  const float* bhh = (const float*)d_in[15];
  float* out = (float*)d_out;

  _Float16* p = (_Float16*)d_ws;
  auto take = [&](size_t n){ _Float16* r = p; p += n; return r; };
  _Float16* win16 = take((size_t)H*IN);
  _Float16* aiw16 = take((size_t)H3*H);
  _Float16* aow16 = take((size_t)H*H);
  _Float16* wg16  = take((size_t)H*H3);
  _Float16* wih16 = take((size_t)H3*IN);
  _Float16* whh16 = take((size_t)H3*H);
  _Float16* x16   = take((size_t)B*IN);
  _Float16* sp16  = take((size_t)2*B*H);
  _Float16* xp16  = take((size_t)B*H);
  _Float16* qkv16 = take((size_t)2*B*H3);
  _Float16* ctx16 = take((size_t)2*B*H);
  _Float16* spc16 = take((size_t)B*H);
  _Float16* gin16 = take((size_t)B*H3);
  _Float16* enh16 = take((size_t)B*H);
  _Float16* gi16  = take((size_t)B*H3);
  _Float16* gh16  = take((size_t)B*H3);

  auto cvt = [&](const float* s, _Float16* d, int n){
    k_cvt<<<(n + 255)/256, 256, 0, stream>>>(s, d, n);
  };
  cvt(Win, win16, H*IN);
  cvt(Aiw, aiw16, H3*H);
  cvt(Aow, aow16, H*H);
  cvt(Wg,  wg16,  H*H3);
  cvt(Wih, wih16, H3*IN);
  cvt(Whh, whh16, H3*H);
  cvt(x,   x16,   B*IN);
  k_sp<<<(B*H + 255)/256, 256, 0, stream>>>(hl, hu, sp16, B*H, H);

  dim3 blk(256);
  // x_proj = x @ W_in^T + b_in
  k_gemm<0><<<dim3(H/GBN,  B/GBM),     blk, 0, stream>>>(x16,  win16, bin, xp16,  nullptr, nullptr, B,   H,  IN);
  // qkv = sp @ attn_in_w^T + attn_in_b
  k_gemm<0><<<dim3(H3/GBN, (2*B)/GBM), blk, 0, stream>>>(sp16, aiw16, Aib, qkv16, nullptr, nullptr, 2*B, H3, H);
  // per-(b,head) 2x2 attention -> ctx
  k_attn<<<(B*4*32)/256, 256, 0, stream>>>(qkv16, ctx16, B);
  // spatial_combined = mean over positions of (ctx @ attn_out_w^T + b)
  k_gemm<1><<<dim3(H/GBN,  (2*B)/GBM), blk, 0, stream>>>(ctx16, aow16, Aob, spc16, nullptr, nullptr, 2*B, H,  H);
  // gate input concat
  k_concat<<<(B*H + 255)/256, 256, 0, stream>>>(hp, spc16, xp16, gin16, B*H, H);
  // enhanced_prev = hp + sigmoid(gate_in @ W_gate^T + b_gate) * spatial_combined
  k_gemm<2><<<dim3(H/GBN,  B/GBM),     blk, 0, stream>>>(gin16, wg16, bg, enh16, hp, spc16, B, H, H3);
  // GRU matmuls
  k_gemm<0><<<dim3(H3/GBN, B/GBM),     blk, 0, stream>>>(x16,   wih16, bih, gi16, nullptr, nullptr, B, H3, IN);
  k_gemm<0><<<dim3(H3/GBN, B/GBM),     blk, 0, stream>>>(enh16, whh16, bhh, gh16, nullptr, nullptr, B, H3, H);
  // final GRU nonlinearity
  k_gru<<<(B*H + 255)/256, 256, 0, stream>>>(gi16, gh16, enh16, out, B*H);
}
